// SwinTransformerBlock2D_86311662780642
// MI455X (gfx1250) — matrix operand order repaired and checked by reading, compiled, not run on hardware
//
#include <hip/hip_runtime.h>
#include <hip/hip_bf16.h>

typedef __bf16 bf16_t;
typedef __attribute__((ext_vector_type(8)))  __bf16 v8bf;
typedef __attribute__((ext_vector_type(16))) __bf16 v16bf;
typedef __attribute__((ext_vector_type(8)))  float  v8f;

#define DIMC 384
#define NHD 12
#define HD 32
#define NTOK 49
#define NROWS 50176   // 16*56*56 = 1024*49

__device__ __forceinline__ v8f wmma_bf16(v16bf a, v16bf b, v8f c) {
    return __builtin_amdgcn_wmma_f32_16x16x32_bf16(false, a, false, b, (short)0, c, false, false);
}

__device__ __forceinline__ v16bf ld_frag_split(const bf16_t* lo, const bf16_t* hi) {
    v8bf a = *(const v8bf*)lo;
    v8bf b = *(const v8bf*)hi;
    v16bf r;
#pragma unroll
    for (int i = 0; i < 8; ++i) { r[i] = a[i]; r[i + 8] = b[i]; }
    return r;
}

// ---------------------------------------------------------------------------
// Weight prep: W[K][N] f32  ->  Wt[N][K] bf16 (column-major; done once, tiny).
// Coalesced writes; strided reads are L2-trivial at these sizes.
// ---------------------------------------------------------------------------
__global__ __launch_bounds__(256)
void transpose_w(const float* __restrict__ W, bf16_t* __restrict__ Wt,
                 int K, int N) {
    int idx = blockIdx.x * 256 + threadIdx.x;
    if (idx < K * N) {
        int n = idx / K, k = idx % K;
        Wt[idx] = (bf16_t)W[(size_t)k * N + n];
    }
}

// ---------------------------------------------------------------------------
// Kernel 1: LN1 + cyclic shift (-3,-3) + window partition + bf16 cast.
// One wave (32 lanes) per token; 8 tokens per 256-thread block.
// ---------------------------------------------------------------------------
__global__ __launch_bounds__(256)
void ln1_shift_partition(const float* __restrict__ x,
                         const float* __restrict__ gw,
                         const float* __restrict__ gb,
                         bf16_t* __restrict__ xw) {
    int token = blockIdx.x * 8 + (threadIdx.x >> 5);
    int lane  = threadIdx.x & 31;
    int b_ = token / NTOK, n = token % NTOK;
    int b  = b_ >> 6, wi = b_ & 63;
    int wh = wi >> 3, ww = wi & 7;
    int ph = n / 7,  pw = n % 7;
    int sh = (wh * 7 + ph + 3) % 56;   // shifted source pixel
    int sw = (ww * 7 + pw + 3) % 56;
    const float* src = x + (((size_t)b * 56 + sh) * 56 + sw) * DIMC;

    float v[12], s = 0.f, ss = 0.f;
#pragma unroll
    for (int j = 0; j < 12; ++j) {
        float t = src[lane + 32 * j];
        v[j] = t; s += t; ss += t * t;
    }
#pragma unroll
    for (int d = 16; d; d >>= 1) {
        s  += __shfl_xor(s,  d, 32);
        ss += __shfl_xor(ss, d, 32);
    }
    float mean = s * (1.f / DIMC);
    float rinv = rsqrtf(ss * (1.f / DIMC) - mean * mean + 1e-5f);
    bf16_t* dst = xw + (size_t)token * DIMC;
#pragma unroll
    for (int j = 0; j < 12; ++j) {
        int c = lane + 32 * j;
        dst[c] = (bf16_t)((v[j] - mean) * rinv * gw[c] + gb[c]);
    }
}

// ---------------------------------------------------------------------------
// LN2 (plain rows) -> bf16
// ---------------------------------------------------------------------------
__global__ __launch_bounds__(256)
void ln2_rows(const float* __restrict__ x2,
              const float* __restrict__ gw,
              const float* __restrict__ gb,
              bf16_t* __restrict__ h) {
    int row  = blockIdx.x * 8 + (threadIdx.x >> 5);
    int lane = threadIdx.x & 31;
    const float* src = x2 + (size_t)row * DIMC;
    float v[12], s = 0.f, ss = 0.f;
#pragma unroll
    for (int j = 0; j < 12; ++j) {
        float t = src[lane + 32 * j];
        v[j] = t; s += t; ss += t * t;
    }
#pragma unroll
    for (int d = 16; d; d >>= 1) {
        s  += __shfl_xor(s,  d, 32);
        ss += __shfl_xor(ss, d, 32);
    }
    float mean = s * (1.f / DIMC);
    float rinv = rsqrtf(ss * (1.f / DIMC) - mean * mean + 1e-5f);
    bf16_t* dst = h + (size_t)row * DIMC;
#pragma unroll
    for (int j = 0; j < 12; ++j) {
        int c = lane + 32 * j;
        dst[c] = (bf16_t)((v[j] - mean) * rinv * gw[c] + gb[c]);
    }
}

// ---------------------------------------------------------------------------
// WMMA GEMM: C[M,N] = A[M,K](bf16, row-major) * Wt[N,K](bf16, col-major of W).
// Block tile 128x64, BK=64 (2 WMMA K-steps per stage), 8 waves in a 4x2 grid,
// each wave owns a 32x32 quadrant (2x2 16x16 frags, 8 WMMAs per stage).
// Epilogues: 0 = QKV scatter, 1 = proj + window-reverse + residual,
//            2 = GELU -> bf16, 3 = +bias +residual -> f32 out.
// ---------------------------------------------------------------------------
#define BM 128
#define BN 64
#define BK 64

template <int EPI>
__global__ __launch_bounds__(256)
void gemm_bf16(const bf16_t* __restrict__ A, const bf16_t* __restrict__ Wt,
               const float* __restrict__ bias, const float* __restrict__ addsrc,
               bf16_t* __restrict__ ob0, bf16_t* __restrict__ ob1,
               bf16_t* __restrict__ ob2, float* __restrict__ of,
               int M, int N, int K) {
    __shared__ bf16_t As[BM][BK];   // 16 KB
    __shared__ bf16_t Bt[BN][BK];   // 8 KB (already transposed in global)

    const int tid   = threadIdx.x;
    const int lane  = tid & 31;
    const int wave  = tid >> 5;
    const int wm    = wave >> 1;          // 0..3
    const int wn    = wave & 1;           // 0..1
    const int Mbase = blockIdx.y * BM;
    const int Nbase = blockIdx.x * BN;
    const int khalf = lane >> 4;          // 0/1: which K-half this lane carries
    const int l15   = lane & 15;

    v8f acc[2][2] = {};

    for (int k0 = 0; k0 < K; k0 += BK) {
        // Stage A tile: 128 rows x 64 elts (128 B/row); 2 threads/row, 64 B each.
        {
            int r = tid >> 1, half = tid & 1;
            const uint4* ag = (const uint4*)(A + (size_t)(Mbase + r) * K + k0 + half * 32);
            uint4* as = (uint4*)&As[r][half * 32];
#pragma unroll
            for (int j = 0; j < 4; ++j) as[j] = ag[j];
        }
        // Stage B tile (already bf16 col-major): 64 rows x 64 elts; 4 threads/row.
        {
            int n = tid >> 2, seg = tid & 3;
            const uint4* bg = (const uint4*)(Wt + (size_t)(Nbase + n) * K + k0 + seg * 16);
            uint4* bs = (uint4*)&Bt[n][seg * 16];
            bs[0] = bg[0];
            bs[1] = bg[1];
        }
        if (k0 + BK < K)
            __builtin_prefetch(A + (size_t)(Mbase + (tid >> 1)) * K + k0 + BK, 0, 0);
        __syncthreads();

#pragma unroll
        for (int ks = 0; ks < 2; ++ks) {
            const int kb = ks * 32;
            v16bf af[2], bfg[2];
#pragma unroll
            for (int mi = 0; mi < 2; ++mi) {
                int mrow = wm * 32 + mi * 16 + l15;
                af[mi] = *(const v16bf*)&As[mrow][kb + khalf * 16];
            }
#pragma unroll
            for (int ni = 0; ni < 2; ++ni) {
                int ncol = wn * 32 + ni * 16 + l15;
                bfg[ni] = *(const v16bf*)&Bt[ncol][kb + khalf * 16];
            }
#pragma unroll
            for (int mi = 0; mi < 2; ++mi)
#pragma unroll
                for (int ni = 0; ni < 2; ++ni)
                    acc[mi][ni] = wmma_bf16(af[mi], bfg[ni], acc[mi][ni]);
        }
        __syncthreads();
    }

    // Epilogue. C layout: VGPR r, lanes 0-15 -> M=r, lanes 16-31 -> M=r+8; N=lane&15.
    const int rowoff = khalf << 3;
#pragma unroll
    for (int mi = 0; mi < 2; ++mi) {
#pragma unroll
        for (int ni = 0; ni < 2; ++ni) {
#pragma unroll
            for (int rr = 0; rr < 8; ++rr) {
                int grow = Mbase + wm * 32 + mi * 16 + rr + rowoff;
                int gcol = Nbase + wn * 32 + ni * 16 + l15;
                float val = acc[mi][ni][rr];
                if constexpr (EPI == 0) {  // QKV: scatter to [B_,NH,49,32]
                    int b_ = grow / NTOK, n = grow % NTOK;
                    int which = gcol / DIMC, rem = gcol % DIMC;
                    int head = rem >> 5, dd = rem & 31;
                    size_t o = (((size_t)b_ * NHD + head) * NTOK + n) * HD + dd;
                    float vv = val + bias[gcol];
                    if (which == 0)      ob0[o] = (bf16_t)(vv * 0.17677669529663687f);
                    else if (which == 1) ob1[o] = (bf16_t)vv;
                    else                 ob2[o] = (bf16_t)vv;
                } else if constexpr (EPI == 1) {  // proj: reverse window+shift, +x
                    int b_ = grow / NTOK, n = grow % NTOK;
                    int b = b_ >> 6, wi = b_ & 63;
                    int wh = wi >> 3, ww = wi & 7;
                    int ph = n / 7, pw = n % 7;
                    int fh = (wh * 7 + ph + 3) % 56;
                    int fw = (ww * 7 + pw + 3) % 56;
                    size_t o = (((size_t)b * 56 + fh) * 56 + fw) * DIMC + gcol;
                    of[o] = val + bias[gcol] + addsrc[o];
                } else if constexpr (EPI == 2) {  // GELU (exact) -> bf16
                    float t = val + bias[gcol];
                    float g = 0.5f * t * (1.0f + erff(t * 0.70710678118654752f));
                    ob0[(size_t)grow * N + gcol] = (bf16_t)g;
                } else {  // MLP2: +bias +x2 -> f32
                    size_t o = (size_t)grow * N + gcol;
                    of[o] = val + bias[gcol] + addsrc[o];
                }
            }
        }
    }
}

// ---------------------------------------------------------------------------
// Windowed attention: one block per (window, head). 4 waves; 49 padded to 64.
// S = q*k^T (+relbias +shiftmask), softmax rows, out = P*v -> [B_,49,384] bf16
// ---------------------------------------------------------------------------
__global__ __launch_bounds__(128)
void win_attention(const bf16_t* __restrict__ Q, const bf16_t* __restrict__ Km,
                   const bf16_t* __restrict__ V, const float* __restrict__ relT,
                   bf16_t* __restrict__ out) {
    __shared__ bf16_t qs[64][32];   // 4 KB (rows padded with zeros)
    __shared__ bf16_t ks[64][32];   // 4 KB (== B^T of K-matrix for QK^T)
    __shared__ bf16_t vt[32][64];   // 4 KB (v transposed: B^T for P@V)
    __shared__ float  S[64][64];    // 16 KB
    __shared__ bf16_t P[64][64];    // 8 KB

    int bh = blockIdx.x;
    int b_ = bh / NHD, h = bh % NHD;
    size_t base = ((size_t)b_ * NHD + h) * NTOK * HD;
    int tid = threadIdx.x;

    for (int idx = tid; idx < 64 * 32; idx += 128) {
        int n = idx >> 5, d = idx & 31;
        bf16_t zq = (bf16_t)0.f, zk = zq, zv = zq;
        if (n < NTOK) {
            zq = Q[base + n * HD + d];
            zk = Km[base + n * HD + d];
            zv = V[base + n * HD + d];
        }
        qs[n][d] = zq; ks[n][d] = zk; vt[d][n] = zv;
    }
    __syncthreads();

    int wave = tid >> 5, lane = tid & 31;
    int m0 = wave * 16;
    int khalf = lane >> 4, l15 = lane & 15;
    int mrow = m0 + l15;

    // S = q k^T  (M=64, N=64, K=32)
    v16bf aq = *(const v16bf*)&qs[mrow][khalf * 16];
    v8f acc[4] = {};
#pragma unroll
    for (int nt = 0; nt < 4; ++nt) {
        v16bf bk = *(const v16bf*)&ks[nt * 16 + l15][khalf * 16];
        acc[nt] = wmma_bf16(aq, bk, acc[nt]);
    }
    int wi = b_ & 63, wh = wi >> 3, ww = wi & 7;
#pragma unroll
    for (int nt = 0; nt < 4; ++nt) {
#pragma unroll
        for (int rr = 0; rr < 8; ++rr) {
            int row = m0 + rr + (khalf << 3);
            int col = nt * 16 + l15;
            float sv = acc[nt][rr];
            if (col >= NTOK) {
                sv = -1e30f;                   // K-dim padding -> softmax zero
            } else if (row < NTOK) {
                int ih = row / 7, iw = row % 7, jh = col / 7, jw = col % 7;
                sv += relT[((ih - jh + 6) * 13 + (iw - jw + 6)) * NHD + h];
                int ghi = wh * 7 + ih, gwi = ww * 7 + iw;
                int ghj = wh * 7 + jh, gwj = ww * 7 + jw;
                int idi = (ghi < 49 ? 0 : (ghi < 53 ? 1 : 2)) * 3 +
                          (gwi < 49 ? 0 : (gwi < 53 ? 1 : 2));
                int idj = (ghj < 49 ? 0 : (ghj < 53 ? 1 : 2)) * 3 +
                          (gwj < 49 ? 0 : (gwj < 53 ? 1 : 2));
                if (idi != idj) sv -= 100.0f;
            }
            S[row][col] = sv;
        }
    }
    __syncthreads();

    // Row softmax (rows 0..63; padded rows produce unused-but-finite values)
    if (tid < 64) {
        float m = -1e30f;
        for (int c = 0; c < NTOK; ++c) m = fmaxf(m, S[tid][c]);
        float s = 0.f;
        for (int c = 0; c < NTOK; ++c) {
            float e = __expf(S[tid][c] - m);
            S[tid][c] = e; s += e;
        }
        float rs = 1.f / s;
        for (int c = 0; c < 64; ++c)
            P[tid][c] = (bf16_t)(c < NTOK ? S[tid][c] * rs : 0.f);
    }
    __syncthreads();

    // out = P v  (M=64, N=32, K=64 -> two K steps)
    v8f acc2[2] = {};
#pragma unroll
    for (int kt = 0; kt < 2; ++kt) {
        v16bf pa = *(const v16bf*)&P[mrow][kt * 32 + khalf * 16];
#pragma unroll
        for (int nt = 0; nt < 2; ++nt) {
            v16bf vb = *(const v16bf*)&vt[nt * 16 + l15][kt * 32 + khalf * 16];
            acc2[nt] = wmma_bf16(pa, vb, acc2[nt]);
        }
    }
#pragma unroll
    for (int nt = 0; nt < 2; ++nt) {
#pragma unroll
        for (int rr = 0; rr < 8; ++rr) {
            int row = m0 + rr + (khalf << 3);
            if (row < NTOK)
                out[((size_t)b_ * NTOK + row) * DIMC + h * HD + nt * 16 + l15] =
                    (bf16_t)acc2[nt][rr];
        }
    }
}

// ---------------------------------------------------------------------------
extern "C" void kernel_launch(void* const* d_in, const int* in_sizes, int n_in,
                              void* d_out, int out_size, void* d_ws, size_t ws_size,
                              hipStream_t stream) {
    const float* x        = (const float*)d_in[0];
    const float* norm1_w  = (const float*)d_in[1];
    const float* norm1_b  = (const float*)d_in[2];
    const float* qkv_w    = (const float*)d_in[3];
    const float* qkv_b    = (const float*)d_in[4];
    const float* rel_tab  = (const float*)d_in[5];
    const float* proj_w   = (const float*)d_in[6];
    const float* proj_b   = (const float*)d_in[7];
    const float* norm2_w  = (const float*)d_in[8];
    const float* norm2_b  = (const float*)d_in[9];
    const float* mlp_w1   = (const float*)d_in[10];
    const float* mlp_b1   = (const float*)d_in[11];
    const float* mlp_w2   = (const float*)d_in[12];
    const float* mlp_b2   = (const float*)d_in[13];

    char* ws = (char*)d_ws;
    const size_t SZ = (size_t)NROWS * DIMC * sizeof(bf16_t);  // 38.5 MB
    bf16_t* xw     = (bf16_t*)(ws);            // LN1/shift/partition output
    bf16_t* qb     = (bf16_t*)(ws + 1 * SZ);
    bf16_t* kb     = (bf16_t*)(ws + 2 * SZ);
    bf16_t* vb     = (bf16_t*)(ws + 3 * SZ);
    bf16_t* attn_o = (bf16_t*)(ws + 4 * SZ);
    float*  x2     = (float*) (ws + 5 * SZ);   // 77 MB (5*SZ .. 7*SZ)
    bf16_t* hbuf   = (bf16_t*)(ws + 4 * SZ);   // reuse attn_o after proj
    bf16_t* hidden = (bf16_t*)(ws);            // reuse xw/q/k/v (154 MB)
    bf16_t* wt_qkv = (bf16_t*)(ws + 7 * SZ);   // bf16 col-major weights (~3.6 MB)
    bf16_t* wt_prj = wt_qkv + (size_t)DIMC * 1152;
    bf16_t* wt_w1  = wt_prj + (size_t)DIMC * DIMC;
    bf16_t* wt_w2  = wt_w1  + (size_t)DIMC * 1536;
    float*  yout   = (float*)d_out;

    // 0. One-time weight convert+transpose to bf16 col-major.
    transpose_w<<<(DIMC * 1152 + 255) / 256, 256, 0, stream>>>(qkv_w, wt_qkv, DIMC, 1152);
    transpose_w<<<(DIMC * DIMC + 255) / 256, 256, 0, stream>>>(proj_w, wt_prj, DIMC, DIMC);
    transpose_w<<<(DIMC * 1536 + 255) / 256, 256, 0, stream>>>(mlp_w1, wt_w1, DIMC, 1536);
    transpose_w<<<(1536 * DIMC + 255) / 256, 256, 0, stream>>>(mlp_w2, wt_w2, 1536, DIMC);

    // 1. LN1 + shift + window partition -> bf16 tokens [50176, 384]
    ln1_shift_partition<<<NROWS / 8, 256, 0, stream>>>(x, norm1_w, norm1_b, xw);

    // 2. QKV GEMM: [50176,384] x [384,1152]
    gemm_bf16<0><<<dim3(1152 / BN, NROWS / BM), 256, 0, stream>>>(
        xw, wt_qkv, qkv_b, nullptr, qb, kb, vb, nullptr, NROWS, 1152, DIMC);

    // 3. Windowed attention (1024 windows x 12 heads)
    win_attention<<<1024 * NHD, 128, 0, stream>>>(qb, kb, vb, rel_tab, attn_o);

    // 4. Proj GEMM + window reverse + unshift + residual -> x2 (f32)
    gemm_bf16<1><<<dim3(DIMC / BN, NROWS / BM), 256, 0, stream>>>(
        attn_o, wt_prj, proj_b, x, nullptr, nullptr, nullptr, x2, NROWS, DIMC, DIMC);

    // 5. LN2 -> bf16
    ln2_rows<<<NROWS / 8, 256, 0, stream>>>(x2, norm2_w, norm2_b, hbuf);

    // 6. MLP1 GEMM + exact GELU -> bf16 hidden [50176, 1536]
    gemm_bf16<2><<<dim3(1536 / BN, NROWS / BM), 256, 0, stream>>>(
        hbuf, wt_w1, mlp_b1, nullptr, hidden, nullptr, nullptr, nullptr,
        NROWS, 1536, DIMC);

    // 7. MLP2 GEMM + bias + residual(x2) -> f32 output
    gemm_bf16<3><<<dim3(DIMC / BN, NROWS / BM), 256, 0, stream>>>(
        hidden, wt_w2, mlp_b2, x2, nullptr, nullptr, nullptr, yout,
        NROWS, DIMC, 1536);
}
